// CEGAN_49460843381555
// MI455X (gfx1250) — matrix operand order repaired and checked by reading, not compiled, not run on hardware
//
#include <hip/hip_runtime.h>
#include <hip/hip_bf16.h>

// ---------------------------------------------------------------------------
// CDNA5 / gfx1250 implementation of the CEGAN reference.
// f16 feature tensors feed v_wmma_f32_16x16x32_f16 out of LDS; contiguous
// tiles are staged with async global->LDS (ASYNCcnt) copies; epilogues
// (softmax over 12 neighbors, layernorm, softplus) are VALU.
// ---------------------------------------------------------------------------

typedef __attribute__((ext_vector_type(16))) _Float16 v16h;
typedef __attribute__((ext_vector_type(8)))  _Float16 v8h;
typedef __attribute__((ext_vector_type(8)))  float    v8f;

#define NCRYS 4096
#define MNBR  12
#define EFEA  64
#define CATW  192          // 2*E + A
#define HFEA  128

__device__ __forceinline__ float softplusf(float x) {
  return (x > 20.f) ? x : log1pf(__expf(x));
}

__device__ __forceinline__ v8f wmma_f16(v16h a, v16h b, v8f c) {
  return __builtin_amdgcn_wmma_f32_16x16x32_f16(false, a, false, b, (short)0, c,
                                                false, false);
}

// Async 16B global -> LDS copy (gfx1250, tracked by ASYNCcnt).
__device__ __forceinline__ void async_copy16(unsigned lds_off, const char* g) {
  asm volatile("global_load_async_to_lds_b128 %0, %1, off"
               :: "v"(lds_off), "v"(g)
               : "memory");
}
__device__ __forceinline__ void wait_async0() {
  asm volatile("s_wait_asynccnt 0x0" ::: "memory");
}

// A fragment: 16x32 f16 per ISA 7.12.2; lane L: row M=L%16, K=klo..klo+7 and
// klo+16..klo+23, klo = (L<16 ? 0 : 8).
__device__ __forceinline__ v16h load_a_frag(const _Float16* base, int stride,
                                            int k0, int lane) {
  const int m   = lane & 15;
  const int klo = (lane < 16) ? 0 : 8;
  const _Float16* p = base + m * stride + k0 + klo;
  union { v16h v; v8h h[2]; } u;
  u.h[0] = *(const v8h*)(p);
  u.h[1] = *(const v8h*)(p + 16);
  return u.v;
}

// B fragment: 32x16 f16; column n = lane%16, K = kb..kb+7 and kb+16..kb+23,
// kb = k0 + (lane<16 ? 0 : 8). B stored transposed (K contiguous).
__device__ __forceinline__ v16h load_b_frag(const _Float16* baseT, int stride,
                                            int k0, int lane) {
  const int n  = lane & 15;
  const int kb = k0 + ((lane < 16) ? 0 : 8);
  const _Float16* p = baseT + n * stride + kb;
  union { v16h v; v8h h[2]; } u;
  u.h[0] = *(const v8h*)(p);
  u.h[1] = *(const v8h*)(p + 16);
  return u.v;
}

// D tile 16x16 f32: lane L holds column n=L%16, rows (L<16 ? v : 8+v).
__device__ __forceinline__ void store_d(float* out, int stride, v8f d, int lane) {
  const int n  = lane & 15;
  const int rb = (lane < 16) ? 0 : 8;
#pragma unroll
  for (int v = 0; v < 8; ++v) out[(rb + v) * stride + n] = d[v];
}

// ---------------------------------------------------------------------------
// Gaussian basis expansion
// ---------------------------------------------------------------------------
__global__ __launch_bounds__(256) void gbf_edge_kernel(
    const float* __restrict__ bond, _Float16* __restrict__ edge0, int total) {
  int idx = blockIdx.x * 256 + threadIdx.x;
  if (idx >= total) return;
  int e = idx & 63, r = idx >> 6;
  float f = 8.0f * (float)e / 63.0f;               // linspace(0,8,64)
  float d = bond[r] - f;
  edge0[idx] = (_Float16)__expf(-d * d * 64.0f);   // gamma = 8/64
}

__global__ __launch_bounds__(256) void gbf_angle_kernel(
    const float* __restrict__ ang, _Float16* __restrict__ ang0, int total) {
  int idx = blockIdx.x * 256 + threadIdx.x;
  if (idx >= total) return;
  int a = idx & 63, r = idx >> 6;
  float f = -1.0f + 2.0f * (float)a / 63.0f;       // linspace(-1,1,64)
  float d = ang[r] - f;
  ang0[idx] = (_Float16)__expf(-d * d * 1024.0f);  // gamma = 2/64
}

// ---------------------------------------------------------------------------
// conv_edge: one block = 16 (n,i) pairs = 192 cat rows of width 192
// ---------------------------------------------------------------------------
#define CE_PAIRS 16
#define CE_ROWS  192
#define OUTSTR   66

static constexpr unsigned CE_SMEM =
    CE_ROWS * CATW * 2 + EFEA * CATW * 2 + CE_ROWS * OUTSTR * 4 +
    (CATW + CE_ROWS + 5 * EFEA + CE_PAIRS * EFEA) * 4;

__global__ __launch_bounds__(256) void conv_edge_kernel(
    const _Float16* __restrict__ edge_in, const _Float16* __restrict__ ang_in,
    const int* __restrict__ nbr,
    const float* __restrict__ att_w, const float* __restrict__ att_b,
    const float* __restrict__ lin_w, const float* __restrict__ lin_b,
    const float* __restrict__ g1, const float* __restrict__ b1,
    const float* __restrict__ g2, const float* __restrict__ b2,
    _Float16* __restrict__ edge_out) {
  extern __shared__ char smem[];
  _Float16* catS = (_Float16*)smem;                  // [192][192]
  _Float16* btS  = catS + CE_ROWS * CATW;            // [64][192] (lin_w^T)
  float* outS  = (float*)(btS + EFEA * CATW);        // [192][66]
  float* attwS = outS + CE_ROWS * OUTSTR;            // [192]
  float* alphaS = attwS + CATW;                      // [192]
  float* lbS = alphaS + CE_ROWS;                     // [64]
  float* g1S = lbS + EFEA;  float* b1S = g1S + EFEA;
  float* g2S = b1S + EFEA;  float* b2S = g2S + EFEA;
  float* sumS = b2S + EFEA;                          // [16][64]

  const int tid  = threadIdx.x;
  const int base = blockIdx.x * CE_PAIRS;

  // async-stage cat = [eij | eik | ang]; every segment is a contiguous 128B row
  {
    const unsigned catOff = (unsigned)(uintptr_t)catS;
    for (int idx = tid; idx < CE_ROWS * 3 * 8; idx += 256) {
      int sub = idx & 7, chunk = idx >> 3;
      int row = chunk / 3, seg = chunk % 3;
      int p = row / MNBR, k = row % MNBR;
      int r = base + p;
      const char* g;
      if (seg == 0)      g = (const char*)(edge_in + (size_t)r * EFEA);
      else if (seg == 1) g = (const char*)(edge_in + ((size_t)nbr[r] * MNBR + k) * EFEA);
      else               g = (const char*)(ang_in + ((size_t)r * MNBR + k) * EFEA);
      async_copy16(catOff + (unsigned)((row * CATW + seg * EFEA) * 2 + sub * 16),
                   g + sub * 16);
    }
  }
  for (int idx = tid; idx < EFEA * CATW; idx += 256) {
    int n = idx / CATW, k = idx % CATW;
    btS[n * CATW + k] = (_Float16)lin_w[k * EFEA + n];
  }
  for (int idx = tid; idx < CATW; idx += 256) attwS[idx] = att_w[idx];
  if (tid < EFEA) {
    lbS[tid] = lin_b[tid];
    g1S[tid] = g1[tid]; b1S[tid] = b1[tid];
    g2S[tid] = g2[tid]; b2S[tid] = b2[tid];
  }
  wait_async0();
  __syncthreads();

  // attention score per row + leaky relu
  if (tid < CE_ROWS) {
    float s = att_b[0];
    const _Float16* row = catS + tid * CATW;
    for (int c = 0; c < CATW; ++c) s += (float)row[c] * attwS[c];
    alphaS[tid] = (s > 0.f) ? s : 0.01f * s;
  }
  __syncthreads();
  // softmax over the k (neighbor) axis
  if (tid < CE_PAIRS) {
    float mx = -1e30f;
    for (int k = 0; k < MNBR; ++k) mx = fmaxf(mx, alphaS[tid * MNBR + k]);
    float e[MNBR], den = 0.f;
    for (int k = 0; k < MNBR; ++k) { e[k] = __expf(alphaS[tid * MNBR + k] - mx); den += e[k]; }
    float inv = 1.f / den;
    for (int k = 0; k < MNBR; ++k) alphaS[tid * MNBR + k] = e[k] * inv;
  }

  // GEMM [192,192]x[192,64]: 12x4 tiles of 16x16, K = 6 chunks of 32.
  // Wave id forced scalar so the tile loop is wave-uniform (EXEC all-ones).
  {
    const int wave = __builtin_amdgcn_readfirstlane(tid >> 5);
    const int lane = tid & 31;
#pragma unroll 1
    for (int t = 0; t < 6; ++t) {
      int tile = wave + t * 8;
      int rt = tile >> 2, ct = tile & 3;
      v8f acc = {};
      const _Float16* ab = catS + rt * 16 * CATW;
      const _Float16* bb = btS + ct * 16 * CATW;
#pragma unroll
      for (int kc = 0; kc < 6; ++kc) {
        v16h a = load_a_frag(ab, CATW, kc * 32, lane);
        v16h b = load_b_frag(bb, CATW, kc * 32, lane);
        acc = wmma_f16(a, b, acc);
      }
      store_d(outS + rt * 16 * OUTSTR + ct * 16, OUTSTR, acc, lane);
    }
  }
  __syncthreads();

  // h = alpha*(gemm+lb); layernorm(bn1); softplus
  if (tid < CE_ROWS) {
    float alpha = alphaS[tid];
    float* row = outS + tid * OUTSTR;
    float mu = 0.f, m2 = 0.f;
    for (int f = 0; f < EFEA; ++f) {
      float h = alpha * (row[f] + lbS[f]);
      row[f] = h; mu += h; m2 += h * h;
    }
    mu *= (1.f / EFEA); m2 = m2 * (1.f / EFEA) - mu * mu;
    float inv = rsqrtf(m2 + 1e-5f);
    for (int f = 0; f < EFEA; ++f)
      row[f] = softplusf((row[f] - mu) * inv * g1S[f] + b1S[f]);
  }
  __syncthreads();

  // residual + sum over k, layernorm(bn2), softplus, writeback (f16)
  if (tid < CE_PAIRS) {
    int r = base + tid;
    float mu = 0.f, m2 = 0.f;
    for (int f = 0; f < EFEA; ++f) {
      float s = (float)catS[(tid * MNBR) * CATW + f];   // eij == edge_in[r]
      for (int k = 0; k < MNBR; ++k) s += outS[(tid * MNBR + k) * OUTSTR + f];
      sumS[tid * EFEA + f] = s; mu += s; m2 += s * s;
    }
    mu *= (1.f / EFEA); m2 = m2 * (1.f / EFEA) - mu * mu;
    float inv = rsqrtf(m2 + 1e-5f);
    for (int f = 0; f < EFEA; ++f) {
      float y = (sumS[tid * EFEA + f] - mu) * inv * g2S[f] + b2S[f];
      edge_out[r * EFEA + f] = (_Float16)softplusf(y);
    }
  }
}

// ---------------------------------------------------------------------------
// conv_angle: same cat construction; no softmax; per-row residual + LN
// ---------------------------------------------------------------------------
static constexpr unsigned CA_SMEM =
    CE_ROWS * CATW * 2 + EFEA * CATW * 2 + CE_ROWS * OUTSTR * 4 +
    (CATW + CE_ROWS + 3 * EFEA) * 4;

__global__ __launch_bounds__(256) void conv_angle_kernel(
    const _Float16* __restrict__ ang_in, const _Float16* __restrict__ edge_in,
    const int* __restrict__ nbr,
    const float* __restrict__ att_w, const float* __restrict__ att_b,
    const float* __restrict__ lin_w, const float* __restrict__ lin_b,
    const float* __restrict__ g2, const float* __restrict__ b2,
    _Float16* __restrict__ ang_out) {
  extern __shared__ char smem[];
  _Float16* catS = (_Float16*)smem;
  _Float16* btS  = catS + CE_ROWS * CATW;
  float* outS  = (float*)(btS + EFEA * CATW);
  float* attwS = outS + CE_ROWS * OUTSTR;
  float* alphaS = attwS + CATW;
  float* lbS = alphaS + CE_ROWS;
  float* g2S = lbS + EFEA; float* b2S = g2S + EFEA;

  const int tid  = threadIdx.x;
  const int base = blockIdx.x * CE_PAIRS;

  {
    const unsigned catOff = (unsigned)(uintptr_t)catS;
    for (int idx = tid; idx < CE_ROWS * 3 * 8; idx += 256) {
      int sub = idx & 7, chunk = idx >> 3;
      int row = chunk / 3, seg = chunk % 3;
      int p = row / MNBR, k = row % MNBR;
      int r = base + p;
      const char* g;
      if (seg == 0)      g = (const char*)(edge_in + (size_t)r * EFEA);
      else if (seg == 1) g = (const char*)(edge_in + ((size_t)nbr[r] * MNBR + k) * EFEA);
      else               g = (const char*)(ang_in + ((size_t)r * MNBR + k) * EFEA);
      async_copy16(catOff + (unsigned)((row * CATW + seg * EFEA) * 2 + sub * 16),
                   g + sub * 16);
    }
  }
  for (int idx = tid; idx < EFEA * CATW; idx += 256) {
    int n = idx / CATW, k = idx % CATW;
    btS[n * CATW + k] = (_Float16)lin_w[k * EFEA + n];
  }
  for (int idx = tid; idx < CATW; idx += 256) attwS[idx] = att_w[idx];
  if (tid < EFEA) { lbS[tid] = lin_b[tid]; g2S[tid] = g2[tid]; b2S[tid] = b2[tid]; }
  wait_async0();
  __syncthreads();

  if (tid < CE_ROWS) {
    float s = att_b[0];
    const _Float16* row = catS + tid * CATW;
    for (int c = 0; c < CATW; ++c) s += (float)row[c] * attwS[c];
    alphaS[tid] = (s > 0.f) ? s : 0.01f * s;   // leaky relu, no softmax
  }

  {
    const int wave = __builtin_amdgcn_readfirstlane(tid >> 5);
    const int lane = tid & 31;
#pragma unroll 1
    for (int t = 0; t < 6; ++t) {
      int tile = wave + t * 8;
      int rt = tile >> 2, ct = tile & 3;
      v8f acc = {};
      const _Float16* ab = catS + rt * 16 * CATW;
      const _Float16* bb = btS + ct * 16 * CATW;
#pragma unroll
      for (int kc = 0; kc < 6; ++kc) {
        v16h a = load_a_frag(ab, CATW, kc * 32, lane);
        v16h b = load_b_frag(bb, CATW, kc * 32, lane);
        acc = wmma_f16(a, b, acc);
      }
      store_d(outS + rt * 16 * OUTSTR + ct * 16, OUTSTR, acc, lane);
    }
  }
  __syncthreads();

  if (tid < CE_ROWS) {
    float alpha = alphaS[tid];
    float* row = outS + tid * OUTSTR;
    float mu = 0.f, m2 = 0.f;
    for (int f = 0; f < EFEA; ++f) {
      float s = (float)catS[tid * CATW + 2 * EFEA + f] + alpha * (row[f] + lbS[f]);
      row[f] = s; mu += s; m2 += s * s;
    }
    mu *= (1.f / EFEA); m2 = m2 * (1.f / EFEA) - mu * mu;
    float inv = rsqrtf(m2 + 1e-5f);
    long gr = (long)base * MNBR + tid;       // global (n,i,k) row
    for (int f = 0; f < EFEA; ++f) {
      float y = (row[f] - mu) * inv * g2S[f] + b2S[f];
      ang_out[gr * EFEA + f] = (_Float16)softplusf(y);
    }
  }
}

// ---------------------------------------------------------------------------
// expand_edge + pool over neighbors: e_pool[n,f] = sum_i softplus(edge@W + b)
// ---------------------------------------------------------------------------
#define HSTR 132
static constexpr unsigned XE_SMEM =
    48 * EFEA * 2 + HFEA * EFEA * 2 + 48 * HSTR * 4 + HFEA * 4;

__global__ __launch_bounds__(256) void expand_edge_kernel(
    const _Float16* __restrict__ edge, const float* __restrict__ w,
    const float* __restrict__ b, float* __restrict__ e_pool) {
  extern __shared__ char smem[];
  _Float16* eS  = (_Float16*)smem;             // [48][64]
  _Float16* wtS = eS + 48 * EFEA;              // [128][64]
  float* outS = (float*)(wtS + HFEA * EFEA);   // [48][132]
  float* bS   = outS + 48 * HSTR;              // [128]
  const int tid = threadIdx.x;
  const int n0  = blockIdx.x * 4;              // 4 crystals -> 48 rows

  {
    const unsigned eOff = (unsigned)(uintptr_t)eS;
    const char* gsrc = (const char*)(edge + (size_t)n0 * MNBR * EFEA);
    for (int idx = tid; idx < (48 * EFEA * 2) / 16; idx += 256)   // 384
      async_copy16(eOff + idx * 16, gsrc + idx * 16);
  }
  for (int idx = tid; idx < HFEA * EFEA; idx += 256) {
    int n = idx >> 6, k = idx & 63;
    wtS[n * EFEA + k] = (_Float16)w[k * HFEA + n];
  }
  if (tid < HFEA) bS[tid] = b[tid];
  wait_async0();
  __syncthreads();

  {
    const int wave = __builtin_amdgcn_readfirstlane(tid >> 5);
    const int lane = tid & 31;
#pragma unroll 1
    for (int t = 0; t < 3; ++t) {               // 3 row x 8 col tiles
      int tile = wave + t * 8;
      int rt = tile >> 3, ct = tile & 7;
      v8f acc = {};
#pragma unroll
      for (int kc = 0; kc < 2; ++kc) {
        v16h a = load_a_frag(eS + rt * 16 * EFEA, EFEA, kc * 32, lane);
        v16h bb = load_b_frag(wtS + ct * 16 * EFEA, EFEA, kc * 32, lane);
        acc = wmma_f16(a, bb, acc);
      }
      store_d(outS + rt * 16 * HSTR + ct * 16, HSTR, acc, lane);
    }
  }
  __syncthreads();

  for (int idx = tid; idx < 4 * HFEA; idx += 256) {
    int nl = idx >> 7, f = idx & 127;
    float s = 0.f;
    for (int i = 0; i < MNBR; ++i)
      s += softplusf(outS[(nl * MNBR + i) * HSTR + f] + bS[f]);
    e_pool[(long)(n0 + nl) * HFEA + f] = s;
  }
}

// ---------------------------------------------------------------------------
// expand_angle + double pool: one block per crystal (144 rows)
// ---------------------------------------------------------------------------
static constexpr unsigned XA_SMEM =
    144 * EFEA * 2 + HFEA * EFEA * 2 + 144 * HSTR * 4 + MNBR * HFEA * 4 + HFEA * 4;

__global__ __launch_bounds__(256) void expand_angle_kernel(
    const _Float16* __restrict__ ang, const float* __restrict__ w,
    const float* __restrict__ b, float* __restrict__ a_pool) {
  extern __shared__ char smem[];
  _Float16* aS  = (_Float16*)smem;             // [144][64]
  _Float16* wtS = aS + 144 * EFEA;             // [128][64]
  float* outS = (float*)(wtS + HFEA * EFEA);   // [144][132]
  float* tS   = outS + 144 * HSTR;             // [12][128]
  float* bS   = tS + MNBR * HFEA;              // [128]
  const int tid = threadIdx.x;
  const long n  = blockIdx.x;

  {
    const unsigned aOff = (unsigned)(uintptr_t)aS;
    const char* gsrc = (const char*)(ang + n * 144 * EFEA);
    for (int idx = tid; idx < (144 * EFEA * 2) / 16; idx += 256)  // 1152
      async_copy16(aOff + idx * 16, gsrc + idx * 16);
  }
  for (int idx = tid; idx < HFEA * EFEA; idx += 256) {
    int c = idx >> 6, k = idx & 63;
    wtS[c * EFEA + k] = (_Float16)w[k * HFEA + c];
  }
  if (tid < HFEA) bS[tid] = b[tid];
  wait_async0();
  __syncthreads();

  {
    const int wave = __builtin_amdgcn_readfirstlane(tid >> 5);
    const int lane = tid & 31;
#pragma unroll 1
    for (int t = 0; t < 9; ++t) {               // 9 row x 8 col tiles
      int tile = wave + t * 8;
      int rt = tile >> 3, ct = tile & 7;
      v8f acc = {};
#pragma unroll
      for (int kc = 0; kc < 2; ++kc) {
        v16h a = load_a_frag(aS + rt * 16 * EFEA, EFEA, kc * 32, lane);
        v16h bb = load_b_frag(wtS + ct * 16 * EFEA, EFEA, kc * 32, lane);
        acc = wmma_f16(a, bb, acc);
      }
      store_d(outS + rt * 16 * HSTR + ct * 16, HSTR, acc, lane);
    }
  }
  __syncthreads();

  for (int idx = tid; idx < MNBR * HFEA; idx += 256) {
    int i = idx >> 7, f = idx & 127;
    float s = 0.f;
    for (int k = 0; k < MNBR; ++k)
      s += softplusf(outS[(i * MNBR + k) * HSTR + f] + bS[f]);
    tS[idx] = s;
  }
  __syncthreads();
  if (tid < HFEA) {
    float s = 0.f;
    for (int i = 0; i < MNBR; ++i) s += softplusf(tS[i * HFEA + tid]);
    a_pool[n * HFEA + tid] = s;
  }
}

// ---------------------------------------------------------------------------
// final: concat pools, layernorm over 256, softplus, @ out_w[256,2] + out_b
// ---------------------------------------------------------------------------
__global__ __launch_bounds__(256) void final_kernel(
    const float* __restrict__ e_pool, const float* __restrict__ a_pool,
    const float* __restrict__ g, const float* __restrict__ b,
    const float* __restrict__ ow, const float* __restrict__ ob,
    float* __restrict__ out) {
  __shared__ float r0[256];
  __shared__ float r1[256];
  const int n = blockIdx.x, tid = threadIdx.x;
  float v = (tid < HFEA) ? e_pool[(long)n * HFEA + tid]
                         : a_pool[(long)n * HFEA + (tid - HFEA)];
  r0[tid] = v; __syncthreads();
  for (int s = 128; s > 0; s >>= 1) { if (tid < s) r0[tid] += r0[tid + s]; __syncthreads(); }
  float mu = r0[0] * (1.f / 256.f); __syncthreads();
  float d = v - mu;
  r0[tid] = d * d; __syncthreads();
  for (int s = 128; s > 0; s >>= 1) { if (tid < s) r0[tid] += r0[tid + s]; __syncthreads(); }
  float var = r0[0] * (1.f / 256.f); __syncthreads();
  float y = softplusf(d * rsqrtf(var + 1e-5f) * g[tid] + b[tid]);
  r0[tid] = y * ow[tid * 2 + 0];
  r1[tid] = y * ow[tid * 2 + 1];
  __syncthreads();
  for (int s = 128; s > 0; s >>= 1) {
    if (tid < s) { r0[tid] += r0[tid + s]; r1[tid] += r1[tid + s]; }
    __syncthreads();
  }
  if (tid == 0) {
    out[(long)n * 2 + 0] = r0[0] + ob[0];
    out[(long)n * 2 + 1] = r1[0] + ob[1];
  }
}

// ---------------------------------------------------------------------------
extern "C" void kernel_launch(void* const* d_in, const int* in_sizes, int n_in,
                              void* d_out, int out_size, void* d_ws, size_t ws_size,
                              hipStream_t stream) {
  (void)in_sizes; (void)n_in; (void)out_size; (void)ws_size;
  const float* bond  = (const float*)d_in[0];
  const float* angle = (const float*)d_in[1];
  const int*   nbr   = (const int*)d_in[2];
  // d_in[3] = crys_idx (unused by the reference computation)
  int idx = 4;
  struct EdgeP { const float *aw,*ab,*lw,*lb,*g1,*b1,*g2,*b2; } ep[3];
  for (int i = 0; i < 3; ++i) {
    ep[i].aw = (const float*)d_in[idx++]; ep[i].ab = (const float*)d_in[idx++];
    ep[i].lw = (const float*)d_in[idx++]; ep[i].lb = (const float*)d_in[idx++];
    ep[i].g1 = (const float*)d_in[idx++]; ep[i].b1 = (const float*)d_in[idx++];
    ep[i].g2 = (const float*)d_in[idx++]; ep[i].b2 = (const float*)d_in[idx++];
  }
  struct AngP { const float *aw,*ab,*lw,*lb,*g2,*b2; } ap[2];
  for (int i = 0; i < 2; ++i) {
    ap[i].aw = (const float*)d_in[idx++]; ap[i].ab = (const float*)d_in[idx++];
    ap[i].lw = (const float*)d_in[idx++]; ap[i].lb = (const float*)d_in[idx++];
    ap[i].g2 = (const float*)d_in[idx++]; ap[i].b2 = (const float*)d_in[idx++];
  }
  const float* ee_w = (const float*)d_in[idx++];
  const float* ee_b = (const float*)d_in[idx++];
  const float* ea_w = (const float*)d_in[idx++];
  const float* ea_b = (const float*)d_in[idx++];
  const float* bn_g = (const float*)d_in[idx++];
  const float* bn_b = (const float*)d_in[idx++];
  const float* ow   = (const float*)d_in[idx++];
  const float* ob   = (const float*)d_in[idx++];

  const long NE = (long)NCRYS * MNBR;            // 49152 edge rows
  const long NA = NE * MNBR;                     // 589824 angle rows

  char* ws = (char*)d_ws;
  _Float16* edgeA = (_Float16*)ws;
  _Float16* edgeB = edgeA + NE * EFEA;
  _Float16* angA  = edgeB + NE * EFEA;
  _Float16* angB  = angA + NA * EFEA;
  float* e_pool = (float*)(angB + NA * EFEA);
  float* a_pool = e_pool + (long)NCRYS * HFEA;

  // GBF expansion to f16 features
  {
    long t = NE * EFEA;
    gbf_edge_kernel<<<dim3((t + 255) / 256), dim3(256), 0, stream>>>(bond, edgeA, (int)t);
    t = NA * EFEA;
    gbf_angle_kernel<<<dim3((t + 255) / 256), dim3(256), 0, stream>>>(angle, angA, (int)t);
  }

  const int convGrid = (int)(NE / CE_PAIRS);     // 3072 blocks

  // conv stack (ping-pong through workspace)
  conv_edge_kernel<<<convGrid, 256, CE_SMEM, stream>>>(
      edgeA, angA, nbr, ep[0].aw, ep[0].ab, ep[0].lw, ep[0].lb,
      ep[0].g1, ep[0].b1, ep[0].g2, ep[0].b2, edgeB);
  conv_angle_kernel<<<convGrid, 256, CA_SMEM, stream>>>(
      angA, edgeB, nbr, ap[0].aw, ap[0].ab, ap[0].lw, ap[0].lb,
      ap[0].g2, ap[0].b2, angB);
  conv_edge_kernel<<<convGrid, 256, CE_SMEM, stream>>>(
      edgeB, angB, nbr, ep[1].aw, ep[1].ab, ep[1].lw, ep[1].lb,
      ep[1].g1, ep[1].b1, ep[1].g2, ep[1].b2, edgeA);
  conv_angle_kernel<<<convGrid, 256, CA_SMEM, stream>>>(
      angB, edgeA, nbr, ap[1].aw, ap[1].ab, ap[1].lw, ap[1].lb,
      ap[1].g2, ap[1].b2, angA);
  conv_edge_kernel<<<convGrid, 256, CE_SMEM, stream>>>(
      edgeA, angA, nbr, ep[2].aw, ep[2].ab, ep[2].lw, ep[2].lb,
      ep[2].g1, ep[2].b1, ep[2].g2, ep[2].b2, edgeB);

  // expand + pool
  expand_edge_kernel<<<NCRYS / 4, 256, XE_SMEM, stream>>>(edgeB, ee_w, ee_b, e_pool);
  expand_angle_kernel<<<NCRYS, 256, XA_SMEM, stream>>>(angA, ea_w, ea_b, a_pool);

  // final layernorm + classifier
  final_kernel<<<NCRYS, 256, 0, stream>>>(e_pool, a_pool, bn_g, bn_b, ow, ob,
                                          (float*)d_out);
}
